// MoEMLP_50646254355256
// MI455X (gfx1250) — compile-verified, matrix operand order checked by reading, not run on hardware
//
#include <hip/hip_runtime.h>

typedef __attribute__((ext_vector_type(16))) _Float16 v16h;
typedef __attribute__((ext_vector_type(8)))  _Float16 v8h;
typedef __attribute__((ext_vector_type(8)))  float    v8f;
typedef __attribute__((ext_vector_type(4)))  int      v4i;

#define HIDDEN 1024
#define INTER  4096
#define NE     8
#define MAXTOK 2048
#define ALPHA  1.702f
#define LIMIT  7.0f

// Pre-dequantized fragment pools (dwords = packed 2 x f16)
// gup: [E][INTER/16][HIDDEN/32][2 (gate,up)][32 lanes][8 dw]  = 33,554,432 dw (128 MB)
// dn : [E][HIDDEN/16][INTER/32][32 lanes][8 dw]               = 16,777,216 dw ( 64 MB)
#define FRAG_GUP_DWORDS (NE * (INTER / 16) * (HIDDEN / 32) * 2 * 32 * 8)
#define FRAG_DN_DWORDS  (NE * (HIDDEN / 16) * (INTER / 32) * 32 * 8)

union FragH {
    v16h v;
    v8h  half2[2];
    v4i  w4[2];
    unsigned int w[8];
};

// ---------------------------------------------------------------------------
// MXFP4 dequant (used only in the one-shot pre-pass).
// f16 bits of the 8 FP4 magnitudes {0,.5,1,1.5,2,3,4,6} have a zero low byte;
// high bytes form an 8-entry byte LUT in two dwords. The E8M0 scale is a
// power of two: +shift on the f16 exponent == +shift<<2 on the LUT byte, and
// for shift in [-9,+5] every nonzero entry stays in [0x14,0x5A] (no byte
// carry), entry 0 stays 0 so zeros stay exact.
// ---------------------------------------------------------------------------
__device__ __forceinline__ void fp4_tables(int scale_e8m0,
                                           unsigned int& tbl_lo,
                                           unsigned int& tbl_hi) {
    unsigned int s = (unsigned int)(scale_e8m0 - 127);
    tbl_lo = 0x3E3C3800u + 0x04040400u * s;   // mags 0..3 (entry 0 stays 0x00)
    tbl_hi = 0x46444240u + 0x04040404u * s;   // mags 4..7
}

// q = one byte (low nibble = even element, high = odd) -> 2 packed f16
__device__ __forceinline__ unsigned int fp4x2(unsigned int q,
                                              unsigned int tbl_lo,
                                              unsigned int tbl_hi) {
    unsigned int sel = (q & 7u) | (((q >> 4) & 7u) << 8);      // sel bytes 0,1
    unsigned int r = __builtin_amdgcn_perm(tbl_hi, tbl_lo, sel);
    unsigned int d = ((r << 8) & 0x0000FF00u) | ((r << 16) & 0xFF000000u);
    d |= ((q & 8u) << 12) | ((q & 0x80u) << 24);               // sign bits
    return d;
}

// Byte index inside a 16-byte MXFP4 block for fragment dword t, K-half h:
//   t<4 -> 4h+t (K = 8h+2t,8h+2t+1)    t>=4 -> 4+4h+t (K = 16+8h+...)
__device__ __forceinline__ int frag_byte_idx(int t, int h) {
    return (t < 4) ? (4 * h + t) : (4 + 4 * h + t);
}

// ------------- Pre-pass A: dequant+swizzle gate_up weights ------------------
__global__ __launch_bounds__(256) void dequant_gup(
        const int* __restrict__ blocks,   // [E, 2I, H/32, 16]
        const int* __restrict__ scales,   // [E, 2I, H/32]
        unsigned int* __restrict__ out) {
    int g = blockIdx.x * 256 + threadIdx.x;          // == output dword index
    int t    = g & 7;
    int lane = (g >> 3) & 31;
    int s    = (g >> 8) & 1;                          // 0=gate row, 1=up row
    int b    = (g >> 9) & 31;                         // K-block
    int jt   = (g >> 14) & 255;                       // 16-channel tile
    int e    = g >> 22;
    int n = lane & 15, h = lane >> 4;
    int row = 2 * (jt * 16 + n) + s;                  // interleaved gate/up
    size_t blk = (size_t)(e * (2 * INTER) + row) * (HIDDEN / 32) + b;
    unsigned int q = (unsigned int)blocks[blk * 16 + frag_byte_idx(t, h)];
    unsigned int tbl_lo, tbl_hi;
    fp4_tables(scales[blk], tbl_lo, tbl_hi);
    out[g] = fp4x2(q, tbl_lo, tbl_hi);
}

// ------------- Pre-pass B: dequant+swizzle down weights ---------------------
__global__ __launch_bounds__(256) void dequant_dn(
        const int* __restrict__ blocks,   // [E, H, I/32, 16]
        const int* __restrict__ scales,   // [E, H, I/32]
        unsigned int* __restrict__ out) {
    int g = blockIdx.x * 256 + threadIdx.x;
    int t    = g & 7;
    int lane = (g >> 3) & 31;
    int b    = (g >> 8) & 127;                        // K-block over INTER
    int dt   = (g >> 15) & 63;                        // 16-output tile
    int e    = g >> 21;
    int n = lane & 15, h = lane >> 4;
    int row = dt * 16 + n;
    size_t blk = (size_t)(e * HIDDEN + row) * (INTER / 32) + b;
    unsigned int q = (unsigned int)blocks[blk * 16 + frag_byte_idx(t, h)];
    unsigned int tbl_lo, tbl_hi;
    fp4_tables(scales[blk], tbl_lo, tbl_hi);
    out[g] = fp4x2(q, tbl_lo, tbl_hi);
}

// Load a pre-swizzled B fragment: 32 contiguous bytes per lane (2 x b128)
__device__ __forceinline__ v16h load_bfrag(const unsigned int* __restrict__ p) {
    FragH B;
    B.w4[0] = ((const v4i*)p)[0];
    B.w4[1] = ((const v4i*)p)[1];
    return B.v;
}

// Build 16x32 f16 A fragment (lane = row m, hB = K-half) from LDS (2 x b128).
__device__ __forceinline__ v16h build_a(const _Float16* __restrict__ row, int hB) {
    const v8h* p = (const v8h*)row;        // 16B aligned
    FragH A;
    A.half2[0] = p[hB];                    // K = 8h .. 8h+7
    A.half2[1] = p[2 + hB];                // K = 16+8h .. 16+8h+7
    return A.v;
}

__device__ __forceinline__ v8f wmma16(v16h a, v16h b, v8f c) {
    return __builtin_amdgcn_wmma_f32_16x16x32_f16(false, a, false, b, (short)0, c, false, false);
}

// ---------------- Kernel 1: router (logits -> top2 -> softmax -> lists) -----
__global__ void moe_router(const float* __restrict__ x,
                           const float* __restrict__ rw,
                           const float* __restrict__ rb,
                           int* __restrict__ cnt,
                           int* __restrict__ tok_list,
                           float* __restrict__ coef_list,
                           int ntok) {
    int t = blockIdx.x * blockDim.x + threadIdx.x;
    if (t >= ntok) return;
    const float* xr = x + (size_t)t * HIDDEN;
    float logit[NE];
#pragma unroll
    for (int e = 0; e < NE; ++e) {
        const float* w = rw + e * HIDDEN;
        float s = 0.f;
        for (int d = 0; d < HIDDEN; ++d) s = fmaf(xr[d], w[d], s);
        logit[e] = s + rb[e];
    }
    int b0 = 0; float v0 = logit[0];
#pragma unroll
    for (int e = 1; e < NE; ++e)
        if (logit[e] > v0) { v0 = logit[e]; b0 = e; }   // strict > : lowest index wins ties
    int b1 = -1; float v1 = -3.4e38f;
#pragma unroll
    for (int e = 0; e < NE; ++e) {
        if (e == b0) continue;
        if (logit[e] > v1) { v1 = logit[e]; b1 = e; }
    }
    float e1 = __expf(v1 - v0);          // v1 <= v0, safe
    float den = 1.f + e1;
    float w0 = 1.f / den;
    float w1 = e1 / den;
    int p0 = atomicAdd(&cnt[b0], 1);
    tok_list[b0 * MAXTOK + p0] = t;
    coef_list[b0 * MAXTOK + p0] = w0;
    int p1 = atomicAdd(&cnt[b1], 1);
    tok_list[b1 * MAXTOK + p1] = t;
    coef_list[b1 * MAXTOK + p1] = w1;
}

// ---------------- Kernel 2: fused expert MLP over 16-token tiles ------------
__global__ __launch_bounds__(256) void moe_expert(
        const float* __restrict__ x,
        const unsigned int* __restrict__ frag_gup,   // pre-dequantized fragments
        const unsigned int* __restrict__ frag_dn,
        const float* __restrict__ gup_bias,          // [E, 2I]
        const float* __restrict__ dn_bias,           // [E, H]
        const int*   __restrict__ cnt,
        const int*   __restrict__ tok_list,
        const float* __restrict__ coef_list,
        float* __restrict__ out) {
    extern __shared__ char smem[];
    _Float16* lds_x   = (_Float16*)smem;                          // 16 x 1024 f16 = 32 KB
    _Float16* lds_act = (_Float16*)(smem + 16 * HIDDEN * 2);      // 16 x 4096 f16 = 128 KB
    int*   s_tok  = (int*)(smem + 16 * HIDDEN * 2 + 16 * INTER * 2);
    float* s_coef = (float*)(s_tok + 16);

    const int e  = blockIdx.y;
    const int ce = cnt[e];
    const int t0 = blockIdx.x * 16;
    if (t0 >= ce) return;          // uniform per block

    const int tid = threadIdx.x;
    if (tid < 16) {
        int idx = t0 + tid;
        bool valid = idx < ce;
        s_tok[tid]  = valid ? tok_list[e * MAXTOK + idx] : -1;
        s_coef[tid] = valid ? coef_list[e * MAXTOK + idx] : 0.f;
    }
    __syncthreads();

    // Stage X tile (f32 -> f16), zero-pad tail rows
    for (int i = tid; i < 16 * HIDDEN; i += 256) {
        int m = i >> 10, d = i & (HIDDEN - 1);
        int tk = s_tok[m];
        float v = (tk >= 0) ? x[(size_t)tk * HIDDEN + d] : 0.f;
        lds_x[m * HIDDEN + d] = (_Float16)v;
    }
    __syncthreads();

    const int lane = tid & 31;
    const int wave = tid >> 5;
    const int nB = lane & 15;      // C/D column
    const int hB = lane >> 4;      // K-half / M-half selector

    // ---- Phase 1: gate_up GEMM + activation -> lds_act ----
    for (int jt = wave; jt < INTER / 16; jt += 8) {
        const int j0 = jt * 16;
        // fragment stream for this (e, jt): [32 kblocks][2][32 lanes][8 dw]
        const unsigned int* fj =
            frag_gup + ((size_t)(e * (INTER / 16) + jt) * (HIDDEN / 32)) * 2 * 256 + lane * 8;
        v8f accg = {}; v8f accu = {};
        const _Float16* xrow = lds_x + nB * HIDDEN;
        for (int b = 0; b < HIDDEN / 32; ++b) {       // K=32 per step
            const unsigned int* pg = fj + (size_t)(b * 2) * 256;
            __builtin_prefetch((const void*)(pg + 512), 0, 3);
            v16h A  = build_a(xrow + b * 32, hB);
            v16h Bg = load_bfrag(pg);
            v16h Bu = load_bfrag(pg + 256);
            accg = wmma16(A, Bg, accg);
            accu = wmma16(A, Bu, accu);
        }
#pragma unroll
        for (int v = 0; v < 8; ++v) {                 // C/D: M = v + 8*h, N = lane%16
            int m  = v + 8 * hB;
            int ch = j0 + nB;
            float g = accg[v] + gup_bias[(size_t)e * 2 * INTER + 2 * ch];
            float u = accu[v] + gup_bias[(size_t)e * 2 * INTER + 2 * ch + 1];
            g = fminf(g, LIMIT);
            u = fminf(fmaxf(u, -LIMIT), LIMIT);
            float sig = 1.f / (1.f + __expf(-ALPHA * g));
            float act = (u + 1.f) * (g * sig);
            lds_act[m * INTER + ch] = (_Float16)act;
        }
    }
    __syncthreads();

    // ---- Phase 2: down GEMM, scale by router coef, atomic-add to output ----
    for (int dt = wave; dt < HIDDEN / 16; dt += 8) {
        const int d0 = dt * 16;
        const unsigned int* fd =
            frag_dn + ((size_t)(e * (HIDDEN / 16) + dt) * (INTER / 32)) * 256 + lane * 8;
        v8f acc = {};
        const _Float16* arow = lds_act + nB * INTER;
        for (int b = 0; b < INTER / 32; ++b) {
            const unsigned int* pd = fd + (size_t)b * 256;
            __builtin_prefetch((const void*)(pd + 256), 0, 3);
            v16h A = build_a(arow + b * 32, hB);
            v16h B = load_bfrag(pd);
            acc = wmma16(A, B, acc);
        }
#pragma unroll
        for (int v = 0; v < 8; ++v) {
            int m  = v + 8 * hB;
            int tk = s_tok[m];
            if (tk >= 0) {
                float val = (acc[v] + dn_bias[(size_t)e * HIDDEN + d0 + nB]) * s_coef[m];
                atomicAdd(&out[(size_t)tk * HIDDEN + d0 + nB], val);
            }
        }
    }
}

extern "C" void kernel_launch(void* const* d_in, const int* in_sizes, int n_in,
                              void* d_out, int out_size, void* d_ws, size_t ws_size,
                              hipStream_t stream) {
    const float* x             = (const float*)d_in[0];
    const float* router_weight = (const float*)d_in[1];
    const float* router_bias   = (const float*)d_in[2];
    const float* gate_up_bias  = (const float*)d_in[3];
    const float* down_bias     = (const float*)d_in[4];
    const int*   gup_blocks    = (const int*)d_in[5];
    const int*   gup_scales    = (const int*)d_in[6];
    const int*   dn_blocks     = (const int*)d_in[7];
    const int*   dn_scales     = (const int*)d_in[8];
    float* out = (float*)d_out;

    const int ntok = in_sizes[0] / HIDDEN;   // 2048

    // Workspace layout: 192 MB fragment pools (== MI455X L2 capacity) + lists
    char* ws = (char*)d_ws;
    unsigned int* frag_gup = (unsigned int*)ws;                               // 128 MB
    unsigned int* frag_dn  = (unsigned int*)(ws + (size_t)FRAG_GUP_DWORDS * 4); // 64 MB
    char* meta = ws + (size_t)(FRAG_GUP_DWORDS + FRAG_DN_DWORDS) * 4;         // ~128.3 KB
    int*   cnt       = (int*)meta;
    int*   tok_list  = (int*)(meta + 256);
    float* coef_list = (float*)(meta + 256 + NE * MAXTOK * 4);

    hipMemsetAsync(cnt, 0, NE * sizeof(int), stream);
    hipMemsetAsync(out, 0, (size_t)out_size * sizeof(float), stream);

    moe_router<<<(ntok + 255) / 256, 256, 0, stream>>>(
        x, router_weight, router_bias, cnt, tok_list, coef_list, ntok);

    dequant_gup<<<FRAG_GUP_DWORDS / 256, 256, 0, stream>>>(gup_blocks, gup_scales, frag_gup);
    dequant_dn <<<FRAG_DN_DWORDS  / 256, 256, 0, stream>>>(dn_blocks,  dn_scales,  frag_dn);

    const size_t lds_bytes = (size_t)16 * HIDDEN * 2 + (size_t)16 * INTER * 2 + 16 * 8;
    hipFuncSetAttribute(reinterpret_cast<const void*>(moe_expert),
                        hipFuncAttributeMaxDynamicSharedMemorySize, (int)lds_bytes);

    dim3 grid((ntok + 15) / 16, NE);
    moe_expert<<<grid, 256, lds_bytes, stream>>>(
        x, frag_gup, frag_dn, gate_up_bias, down_bias,
        cnt, tok_list, coef_list, out);
}
